// MultiHeadAttention_74732430950658
// MI455X (gfx1250) — matrix operand order repaired and checked by reading, not compiled, not run on hardware
//
#include <hip/hip_runtime.h>
#include <hip/hip_bf16.h>

// ---------------------------------------------------------------------------
// MultiHeadAttention for MI455X (gfx1250), bf16 WMMA pipeline, v3:
//  - 16x64 GEMM tile (no spills) with register ping-pong double buffering
//    (no cur<-nxt copies, no scratch)
//  - __launch_bounds__ relaxed occupancy so the allocator keeps frags in VGPRs
//  - attention V-fragment loads hoisted under softmax VALU work
// B=4, T=2048, C=1024, H=16, HD=64
// ---------------------------------------------------------------------------

#define BB 4
#define TT 2048
#define CC 1024
#define HH 16
#define HD 64

typedef __attribute__((ext_vector_type(16))) __bf16 v16bf;
typedef __attribute__((ext_vector_type(8)))  float  v8f;
typedef __attribute__((ext_vector_type(4)))  unsigned int u32x4;

union FragB16 { v16bf v; u32x4 q[2]; };

#define WMMA_BF16(A, Bm, Cm) \
  __builtin_amdgcn_wmma_f32_16x16x32_bf16(false, (A), false, (Bm), (short)0, (Cm), false, false)

__device__ __forceinline__ unsigned short f2bf(float f) {
  unsigned int u = __float_as_uint(f);
  unsigned int r = u + 0x7FFFu + ((u >> 16) & 1u);   // round-to-nearest-even
  return (unsigned short)(r >> 16);
}

// ---------------------------------------------------------------------------
// Stage 1: fp32 -> bf16 conversion (grid-stride)
// ---------------------------------------------------------------------------
__global__ void cvt_f32_to_bf16(const float* __restrict__ src,
                                unsigned short* __restrict__ dst, int n) {
  int i = blockIdx.x * blockDim.x + threadIdx.x;
  int stride = gridDim.x * blockDim.x;
  for (; i < n; i += stride) dst[i] = f2bf(src[i]);
}

// ---------------------------------------------------------------------------
// GEMM core: one wave computes a 16x64 tile of A[M,K] @ W[N,K]^T, K = CC.
// Ping-pong double buffer over K-steps of 32: loads for step k+1 issue before
// the WMMAs of step k, with no inter-buffer copies.
// ---------------------------------------------------------------------------
__device__ __forceinline__ void gc_loadA(FragB16& a, const unsigned short* arow,
                                         int ks, int half) {
  int cbase = ks * 32 + half * 16;
  a.q[0] = *(const u32x4*)(arow + cbase);
  a.q[1] = *(const u32x4*)(arow + cbase + 8);
}

__device__ __forceinline__ void gc_loadB(FragB16 b[4], const unsigned short* wbase,
                                         int obase, int ks, int half, int ln) {
#pragma unroll
  for (int nt = 0; nt < 4; ++nt) {
    const unsigned short* bcol =
        wbase + (size_t)(obase + nt * 16 + ln) * CC + ks * 32 + half * 16;
    b[nt].q[0] = *(const u32x4*)(bcol);
    b[nt].q[1] = *(const u32x4*)(bcol + 8);
  }
}

__device__ __forceinline__ void gemm_core_16x64(const unsigned short* __restrict__ arow,
                                                const unsigned short* __restrict__ wbase,
                                                int obase, int half, int ln,
                                                v8f acc[4]) {
  const int NK = CC / 32;                       // 32 K-steps (even)
  FragB16 a0, a1, b0[4], b1[4];

  gc_loadA(a0, arow, 0, half);
  gc_loadB(b0, wbase, obase, 0, half, ln);

  for (int ks = 0; ks < NK; ks += 2) {
    // issue loads for step ks+1, then compute step ks on buffer 0
    gc_loadA(a1, arow, ks + 1, half);
    gc_loadB(b1, wbase, obase, ks + 1, half, ln);
#pragma unroll
    for (int nt = 0; nt < 4; ++nt) acc[nt] = WMMA_BF16(a0.v, b0[nt].v, acc[nt]);

    // issue loads for step ks+2 (if any), then compute step ks+1 on buffer 1
    if (ks + 2 < NK) {
      gc_loadA(a0, arow, ks + 2, half);
      gc_loadB(b0, wbase, obase, ks + 2, half, ln);
    }
#pragma unroll
    for (int nt = 0; nt < 4; ++nt) acc[nt] = WMMA_BF16(a1.v, b1[nt].v, acc[nt]);
  }
}

// ---------------------------------------------------------------------------
// Stage 2: QKV GEMM.  qkv[M=B*T, N=3C] = x_bf16 @ Wqkv^T, scattered into
// q[B,H,T,HD], k[B,H,T,HD], vt[B,H,HD,T] (V transposed for attention B-frags).
// ---------------------------------------------------------------------------
__global__ void __launch_bounds__(256, 1)
gemm_qkv(const unsigned short* __restrict__ xb,
         const unsigned short* __restrict__ wq,
         unsigned short* __restrict__ qp,
         unsigned short* __restrict__ kp,
         unsigned short* __restrict__ vtp) {
  const int NG = (3 * CC) / 64;                 // 48 n-groups
  int wid  = threadIdx.x >> 5;
  int wg   = blockIdx.x * (blockDim.x >> 5) + wid;
  int mt   = wg / NG;                           // 16-row tile index
  int ng   = wg % NG;                           // 64-col group index
  int lane = threadIdx.x & 31;
  int half = lane >> 4, ln = lane & 15;

  v8f acc[4] = {v8f{}, v8f{}, v8f{}, v8f{}};
  const unsigned short* arow = xb + (size_t)(mt * 16 + ln) * CC;

  gemm_core_16x64(arow, wq, ng * 64, half, ln, acc);

  // Scatter-store epilogue: o = h*192 + j ; j<64 -> Q, j<128 -> K, else V^T.
#pragma unroll
  for (int nt = 0; nt < 4; ++nt) {
    int o = ng * 64 + nt * 16 + ln;
    int h = o / (3 * HD), j = o % (3 * HD);
#pragma unroll
    for (int r = 0; r < 8; ++r) {
      int M  = mt * 16 + r + half * 8;          // global row = b*T + t
      int b_ = M >> 11, t = M & (TT - 1);
      unsigned short val = f2bf(acc[nt][r]);
      if (j < HD)
        qp[(((size_t)(b_ * HH + h) * TT + t) * HD) + j] = val;
      else if (j < 2 * HD)
        kp[(((size_t)(b_ * HH + h) * TT + t) * HD) + (j - HD)] = val;
      else
        vtp[((size_t)(b_ * HH + h) * HD + (j - 2 * HD)) * TT + t] = val;
    }
  }
}

// ---------------------------------------------------------------------------
// Stage 3: causal flash attention.  One wave per (b,h, 16-query block).
// Key blocks of 32.  S via 4 WMMAs, online softmax with half-wave shuffles,
// P restriped through LDS, O += P@V via 4 WMMAs against vt.
// V fragments are loaded BEFORE the softmax so global latency hides under VALU.
// ---------------------------------------------------------------------------
__global__ void __launch_bounds__(128, 1)
attn_flash(const unsigned short* __restrict__ qp,
           const unsigned short* __restrict__ kp,
           const unsigned short* __restrict__ vtp,
           unsigned short* __restrict__ ctx) {
  __shared__ unsigned short plds[4][16 * 32];   // per-wave P tile (bf16)

  int wid  = threadIdx.x >> 5;
  int wg   = blockIdx.x * 4 + wid;
  int qb   = wg & (TT / 16 - 1);                // query block 0..127
  int bh   = wg >> 7;                           // b*H + h
  int lane = threadIdx.x & 31;
  int half = lane >> 4, ln = lane & 15;

  const unsigned short* qbase = qp  + (size_t)bh * TT * HD;
  const unsigned short* kbase = kp  + (size_t)bh * TT * HD;
  const unsigned short* vbase = vtp + (size_t)bh * HD * TT;
  unsigned short* pl = plds[wid];

  // Q fragments are invariant over the key loop: 2 K-steps of 32.
  FragB16 qa[2];
  const unsigned short* qrow = qbase + (size_t)(qb * 16 + ln) * HD;
#pragma unroll
  for (int cs = 0; cs < 2; ++cs) {
    qa[cs].q[0] = *(const u32x4*)(qrow + cs * 32 + half * 16);
    qa[cs].q[1] = *(const u32x4*)(qrow + cs * 32 + half * 16 + 8);
  }

  v8f oacc[4] = {v8f{}, v8f{}, v8f{}, v8f{}};
  float m_st[8], l_st[8];
#pragma unroll
  for (int r = 0; r < 8; ++r) { m_st[r] = -INFINITY; l_st[r] = 0.f; }

  const float scale = 0.125f;                   // HD^-0.5
  int nkb = (qb >> 1) + 1;                      // 32-key blocks up to diagonal

  for (int kb = 0; kb < nkb; ++kb) {
    // ---- S = Q @ K^T (16 x 32 in two accumulators) ----
    v8f s0 = v8f{}, s1 = v8f{};
#pragma unroll
    for (int cs = 0; cs < 2; ++cs) {
      const unsigned short* k0 = kbase + (size_t)(kb * 32 + ln) * HD      + cs * 32 + half * 16;
      const unsigned short* k1 = kbase + (size_t)(kb * 32 + 16 + ln) * HD + cs * 32 + half * 16;
      FragB16 b0, b1;
      b0.q[0] = *(const u32x4*)k0;  b0.q[1] = *(const u32x4*)(k0 + 8);
      b1.q[0] = *(const u32x4*)k1;  b1.q[1] = *(const u32x4*)(k1 + 8);
      s0 = WMMA_BF16(qa[cs].v, b0.v, s0);
      s1 = WMMA_BF16(qa[cs].v, b1.v, s1);
    }

    // ---- issue V fragment loads now; latency hides under softmax VALU ----
    FragB16 vb[4];
#pragma unroll
    for (int dt = 0; dt < 4; ++dt) {
      const unsigned short* vc = vbase + (size_t)(dt * 16 + ln) * TT + kb * 32 + half * 16;
      vb[dt].q[0] = *(const u32x4*)vc;  vb[dt].q[1] = *(const u32x4*)(vc + 8);
    }

    // ---- scale + causal mask (only last block touches the diagonal) ----
    bool lastblk = (kb == nkb - 1);
#pragma unroll
    for (int r = 0; r < 8; ++r) {
      float v0 = s0[r] * scale;
      float v1 = s1[r] * scale;
      if (lastblk) {
        int row = qb * 16 + r + half * 8;
        int k0i = kb * 32 + ln;
        v0 = (k0i      > row) ? -INFINITY : v0;
        v1 = (k0i + 16 > row) ? -INFINITY : v1;
      }
      s0[r] = v0; s1[r] = v1;
    }

    // ---- online softmax: rows live in 16-lane halves of each VGPR ----
#pragma unroll
    for (int r = 0; r < 8; ++r) {
      float mx = fmaxf(s0[r], s1[r]);
      mx = fmaxf(mx, __shfl_xor(mx, 1));
      mx = fmaxf(mx, __shfl_xor(mx, 2));
      mx = fmaxf(mx, __shfl_xor(mx, 4));
      mx = fmaxf(mx, __shfl_xor(mx, 8));
      float mnew  = fmaxf(m_st[r], mx);
      float alpha = __expf(m_st[r] - mnew);
      float p0 = __expf(s0[r] - mnew);
      float p1 = __expf(s1[r] - mnew);
      float rs = p0 + p1;
      rs += __shfl_xor(rs, 1);
      rs += __shfl_xor(rs, 2);
      rs += __shfl_xor(rs, 4);
      rs += __shfl_xor(rs, 8);
      l_st[r] = l_st[r] * alpha + rs;
      m_st[r] = mnew;
      oacc[0][r] *= alpha; oacc[1][r] *= alpha;
      oacc[2][r] *= alpha; oacc[3][r] *= alpha;
      int row = r + half * 8;                   // C-layout -> LDS restripe
      pl[row * 32 + ln]      = f2bf(p0);
      pl[row * 32 + ln + 16] = f2bf(p1);
    }
    asm volatile("s_wait_dscnt 0" ::: "memory");

    // ---- P as A-fragment (16x32), O += P @ V ----
    FragB16 pa;
    pa.q[0] = *(const u32x4*)(pl + ln * 32 + half * 16);
    pa.q[1] = *(const u32x4*)(pl + ln * 32 + half * 16 + 8);
#pragma unroll
    for (int dt = 0; dt < 4; ++dt)
      oacc[dt] = WMMA_BF16(pa.v, vb[dt].v, oacc[dt]);
  }

  // ---- normalize and store context in [B,T,C] (c = h*64 + d), bf16 ----
  int b_ = bh >> 4, h = bh & 15;
#pragma unroll
  for (int r = 0; r < 8; ++r) {
    float inv = 1.0f / l_st[r];
    int t = qb * 16 + r + half * 8;
    size_t base = ((size_t)(b_ * TT + t)) * CC + h * HD;
#pragma unroll
    for (int dt = 0; dt < 4; ++dt)
      ctx[base + dt * 16 + ln] = f2bf(oacc[dt][r] * inv);
  }
}

// ---------------------------------------------------------------------------
// Stage 4: output projection.  out[M=B*T, C] = ctx_bf16 @ Wproj^T + bias, fp32.
// ---------------------------------------------------------------------------
__global__ void __launch_bounds__(256, 1)
gemm_proj(const unsigned short* __restrict__ cb,
          const unsigned short* __restrict__ wp,
          const float* __restrict__ bias,
          float* __restrict__ out) {
  const int NG = CC / 64;                       // 16 n-groups
  int wid  = threadIdx.x >> 5;
  int wg   = blockIdx.x * (blockDim.x >> 5) + wid;
  int mt   = wg / NG;
  int ng   = wg % NG;
  int lane = threadIdx.x & 31;
  int half = lane >> 4, ln = lane & 15;

  v8f acc[4] = {v8f{}, v8f{}, v8f{}, v8f{}};
  const unsigned short* arow = cb + (size_t)(mt * 16 + ln) * CC;

  gemm_core_16x64(arow, wp, ng * 64, half, ln, acc);

#pragma unroll
  for (int nt = 0; nt < 4; ++nt) {
    int o = ng * 64 + nt * 16 + ln;
    float bv = bias[o];
#pragma unroll
    for (int r = 0; r < 8; ++r) {
      int M = mt * 16 + r + half * 8;
      out[(size_t)M * CC + o] = acc[nt][r] + bv;
    }
  }
}

// ---------------------------------------------------------------------------
// Launch
// ---------------------------------------------------------------------------
extern "C" void kernel_launch(void* const* d_in, const int* in_sizes, int n_in,
                              void* d_out, int out_size, void* d_ws, size_t ws_size,
                              hipStream_t stream) {
  const float* x     = (const float*)d_in[0];   // [B,T,C]
  const float* Wqkv  = (const float*)d_in[1];   // [3C,C]
  const float* Wproj = (const float*)d_in[2];   // [C,C]
  const float* bproj = (const float*)d_in[3];   // [C]
  float* out = (float*)d_out;

  unsigned short* ws = (unsigned short*)d_ws;
  size_t off = 0;
  unsigned short* xb = ws + off; off += (size_t)BB * TT * CC;        // x bf16
  unsigned short* wq = ws + off; off += (size_t)3 * CC * CC;         // Wqkv bf16
  unsigned short* wp = ws + off; off += (size_t)CC * CC;             // Wproj bf16
  unsigned short* qp = ws + off; off += (size_t)BB * HH * TT * HD;   // Q
  unsigned short* kp = ws + off; off += (size_t)BB * HH * TT * HD;   // K
  unsigned short* vt = ws + off; off += (size_t)BB * HH * HD * TT;   // V^T
  unsigned short* cb = ws + off; off += (size_t)BB * TT * CC;        // context

  const int nX = BB * TT * CC, nWq = 3 * CC * CC, nWp = CC * CC;
  cvt_f32_to_bf16<<<1024, 256, 0, stream>>>(x, xb, nX);
  cvt_f32_to_bf16<<<1024, 256, 0, stream>>>(Wqkv, wq, nWq);
  cvt_f32_to_bf16<<<1024, 256, 0, stream>>>(Wproj, wp, nWp);

  // QKV GEMM: (8192/16) * (3072/64) = 24576 waves, 8 waves/block
  gemm_qkv<<<3072, 256, 0, stream>>>(xb, wq, qp, kp, vt);

  // Attention: B*H*(T/16) = 8192 waves, 4 waves/block
  attn_flash<<<2048, 128, 0, stream>>>(qp, kp, vt, cb);

  // Projection: (8192/16) * (1024/64) = 8192 waves, 8 waves/block
  gemm_proj<<<1024, 256, 0, stream>>>(cb, wp, bproj, out);
}
